// EntityDisambiguator_27960237097280
// MI455X (gfx1250) — matrix operand order repaired and checked by reading, not compiled, not run on hardware
//
#include <hip/hip_runtime.h>
#include <hip/hip_bf16.h>
#include <math.h>

// ---------------- problem constants (from reference) ----------------
#define PB     32          // batch
#define PT     512         // seq len
#define PS     128         // spans
#define PC     30          // candidates
#define PW     10          // max span width
#define DCTX   768
#define DENT   300
#define DPAD   304         // padded D_ENT stride in workspace
#define PH     100
#define PV     200000

// GEMM tiling
#define BM     64
#define BK     32
#define BN     320         // covers 300 (19 tiles) padded to 20 tiles of 16
#define LDA    34          // bf16 elems per As row (32 + 2 pad, even -> dword aligned)
#define LDB    34          // bf16 elems per Bs row

typedef __bf16 bf16_t;
typedef bf16_t bf16x16 __attribute__((ext_vector_type(16)));
typedef float  floatx8 __attribute__((ext_vector_type(8)));

union FragBF { bf16x16 v; unsigned int d[8]; };

static __device__ __forceinline__ unsigned short f2bf(float f) {
  unsigned int u = __float_as_uint(f);
  u += 0x7FFFu + ((u >> 16) & 1u);     // round-to-nearest-even
  return (unsigned short)(u >> 16);
}

static __device__ __forceinline__ float wave_sum(float v) {
  #pragma unroll
  for (int m = 16; m > 0; m >>= 1) v += __shfl_xor(v, m, 32);
  return v;
}
static __device__ __forceinline__ float wave_max(float v) {
  #pragma unroll
  for (int m = 16; m > 0; m >>= 1) v = fmaxf(v, __shfl_xor(v, m, 32));
  return v;
}

// =====================================================================
// Kernel 1: proj = X(16384x768) @ Wm^T(768x300) + b   via bf16 WMMA
//   grid: 16384/BM = 256 blocks of 256 threads (8 waves).
//   Wave (mw,nw): mw = wave&3 -> 16-row subtile, nw = wave>>2 -> 160-col half.
// =====================================================================
__global__ void __launch_bounds__(256)
proj_gemm_wmma(const float* __restrict__ X, const float* __restrict__ Wm,
               const float* __restrict__ bias, float* __restrict__ proj)
{
  __shared__ unsigned short As[BM * LDA];   // [m][k]  bf16 bits
  __shared__ unsigned short Bs[BN * LDB];   // [n][k]  bf16 bits (k-major per row)

  const int tid  = threadIdx.x;
  const int lane = tid & 31;
  const int wv   = tid >> 5;
  const int mw   = wv & 3;
  const int nw   = wv >> 2;
  const int g    = lane >> 4;     // lane half
  const int ln   = lane & 15;
  const int row0 = blockIdx.x * BM;

  floatx8 acc[10];
  #pragma unroll
  for (int t = 0; t < 10; ++t)
    #pragma unroll
    for (int i = 0; i < 8; ++i) acc[t][i] = 0.0f;

  for (int k0 = 0; k0 < DCTX; k0 += BK) {
    // stage A tile 64x32 (coalesced: 32 consecutive floats per 32 threads)
    for (int i = tid; i < BM * BK; i += 256) {
      int r = i >> 5, c = i & 31;
      As[r * LDA + c] = f2bf(X[(size_t)(row0 + r) * DCTX + k0 + c]);
    }
    // stage B tile: Bs[n][k] = Wm[n*768 + k0 + k], zero-pad n >= 300
    for (int i = tid; i < BN * BK; i += 256) {
      int k = i & 31, n = i >> 5;
      float v = (n < DENT) ? Wm[(size_t)n * DCTX + k0 + k] : 0.0f;
      Bs[n * LDB + k] = f2bf(v);
    }
    __syncthreads();

    // A fragment (16x32 bf16, ISA 7.12.2 layout): lane half g, row ln
    FragBF fa;
    const unsigned short* arow = &As[(mw * 16 + ln) * LDA];
    #pragma unroll
    for (int v = 0; v < 8; ++v) {
      int kb = (v < 4) ? (8 * g + 2 * v) : (16 + 8 * g + 2 * (v - 4));
      fa.d[v] = *(const unsigned int*)(arow + kb);
    }
    #pragma unroll
    for (int nt = 0; nt < 10; ++nt) {
      // B fragment (32x16): same K order as A (8g+s, 16+8g+(s-8)), col ln
      FragBF fb;
      const unsigned short* brow = &Bs[(nw * 160 + nt * 16 + ln) * LDB];
      #pragma unroll
      for (int v = 0; v < 8; ++v)
        fb.d[v] = *(const unsigned int*)(brow + ((v < 4) ? (8 * g + 2 * v) : (16 + 8 * g + 2 * (v - 4))));
      acc[nt] = __builtin_amdgcn_wmma_f32_16x16x32_bf16(
          false, fa.v, false, fb.v, (short)0, acc[nt], false, false);
    }
    __syncthreads();
  }

  // epilogue: C/D layout -> VGPR v holds M = v + 8*g, N = ln
  #pragma unroll
  for (int nt = 0; nt < 10; ++nt) {
    int col = nw * 160 + nt * 16 + ln;
    if (col < DPAD) {
      float bv = (col < DENT) ? bias[col] : 0.0f;
      #pragma unroll
      for (int v = 0; v < 8; ++v) {
        int r = row0 + mw * 16 + v + 8 * g;
        proj[(size_t)r * DPAD + col] = acc[nt][v] + bv;
      }
    }
  }
}

// =====================================================================
// Kernel 2: att_logits[row] = dot(proj[row], attn_W) + attn_b
//   one wave per row; 8 rows per 256-thread block.
// =====================================================================
__global__ void __launch_bounds__(256)
attn_logits_kernel(const float* __restrict__ proj, const float* __restrict__ attW,
                   const float* __restrict__ attb, float* __restrict__ logits)
{
  const int lane = threadIdx.x & 31;
  const int wv   = threadIdx.x >> 5;
  const int row  = blockIdx.x * 8 + wv;
  float p = 0.0f;
  for (int d = lane; d < DENT; d += 32)
    p += proj[(size_t)row * DPAD + d] * attW[d];
  p = wave_sum(p);
  if (lane == 0) logits[row] = p + attb[0];
}

// =====================================================================
// Kernel 3: fused span extractor + entity LN + scoring + softmax + mix
//   one 256-thread block per (b,s). Entity rows read from HBM once,
//   LN'd copies staged in LDS (30x300 f32 = 36 KB of 320 KB/WGP).
// =====================================================================
__global__ void __launch_bounds__(256)
span_entity_fused(const float* __restrict__ proj, const float* __restrict__ logits,
                  const int* __restrict__ spans, const int* __restrict__ cand,
                  const float* __restrict__ priors, const float* __restrict__ etab,
                  const float* __restrict__ kg_g, const float* __restrict__ kg_b,
                  const float* __restrict__ ln_g, const float* __restrict__ ln_b,
                  const float* __restrict__ W1, const float* __restrict__ b1,
                  const float* __restrict__ W2, const float* __restrict__ b2,
                  float* __restrict__ out_link, float* __restrict__ out_wt)
{
  __shared__ float s_ent[PC * DENT];
  __shared__ float s_span[DENT];
  __shared__ float s_attw[16];
  __shared__ int   s_idx[16];
  __shared__ float s_raw[32];
  __shared__ float s_norm[32];
  __shared__ float s_r0[8], s_r1[8];
  __shared__ float s_mu, s_rstd;

  const int tid  = threadIdx.x;
  const int lane = tid & 31;
  const int wv   = tid >> 5;
  const int bs   = blockIdx.x;          // b*S + s
  const int b    = bs >> 7;             // S = 128

  const int start = spans[bs * 2 + 0];
  const int end   = spans[bs * 2 + 1];
  const bool span_valid = (start > -1);
  const int width = end - start;

  // ---- span attention softmax over W positions (wave 0) ----
  if (wv == 0) {
    float lg = -1e30f; int ic = 0;
    if (lane < PW) {
      int idx = start + lane;
      bool valid = (lane <= width) && (idx >= 0) && (idx < PT);
      ic = valid ? idx : 0;
      float raw = logits[b * PT + ic];
      lg = valid ? raw : -1e30f;
    }
    float mx = wave_max(lg);
    float e  = (lane < PW) ? __expf(lg - mx) : 0.0f;
    float sm = wave_sum(e);
    if (lane < PW) { s_attw[lane] = e / sm; s_idx[lane] = ic; }
  }
  __syncthreads();

  // ---- span_rep = sum_w attw[w] * proj[b, idx[w]]; masked ----
  for (int d = tid; d < DENT; d += 256) {
    float a = 0.0f;
    #pragma unroll
    for (int w = 0; w < PW; ++w)
      a += s_attw[w] * proj[(size_t)(b * PT + s_idx[w]) * DPAD + d];
    s_span[d] = span_valid ? a : 0.0f;
  }
  __syncthreads();

  // ---- LayerNorm(span_rep) over 300, block reduction ----
  {
    float ps = 0.0f, pq = 0.0f;
    for (int d = tid; d < DENT; d += 256) { float x = s_span[d]; ps += x; pq += x * x; }
    ps = wave_sum(ps); pq = wave_sum(pq);
    if (lane == 0) { s_r0[wv] = ps; s_r1[wv] = pq; }
    __syncthreads();
    if (tid == 0) {
      float S = 0.0f, Q = 0.0f;
      #pragma unroll
      for (int i = 0; i < 8; ++i) { S += s_r0[i]; Q += s_r1[i]; }
      float mu  = S * (1.0f / DENT);
      float var = Q * (1.0f / DENT) - mu * mu;
      s_mu = mu; s_rstd = rsqrtf(var + 1e-5f);
    }
    __syncthreads();
    float mu = s_mu, rstd = s_rstd;
    for (int d = tid; d < DENT; d += 256)
      s_span[d] = (s_span[d] - mu) * rstd * ln_g[d] + ln_b[d];
  }
  __syncthreads();

  // ---- entity gather + KG LayerNorm + dot scores (one wave per candidate) ----
  const float inv_sqrt_d = 0.05773502691896258f;   // 1/sqrt(300)
  for (int c = wv; c < PC; c += 8) {
    unsigned ue = (unsigned)cand[bs * PC + c];
    if (ue >= (unsigned)PV) ue = 0;
    const float* rowp = etab + (size_t)ue * DENT;
    float ps = 0.0f, pq = 0.0f;
    for (int d = lane; d < DENT; d += 32) {
      float x = rowp[d];
      s_ent[c * DENT + d] = x;
      ps += x; pq += x * x;
    }
    ps = wave_sum(ps); pq = wave_sum(pq);
    float mu   = ps * (1.0f / DENT);
    float rstd = rsqrtf(pq * (1.0f / DENT) - mu * mu + 1e-5f);
    float dot = 0.0f;
    for (int d = lane; d < DENT; d += 32) {
      float x = (s_ent[c * DENT + d] - mu) * rstd * kg_g[d] + kg_b[d];
      s_ent[c * DENT + d] = x;
      dot += x * s_span[d];
    }
    dot = wave_sum(dot);
    if (lane == 0) s_raw[c] = dot * inv_sqrt_d;
  }
  __syncthreads();

  // ---- tiny MLP + mask + softmax over candidates (wave 0, lane = c) ----
  if (wv == 0) {
    const int c = lane;
    float link = -1e38f;
    if (c < PC) {
      float score = s_raw[c];
      float prior = priors[bs * PC + c];
      float a2 = b2[0];
      for (int j = 0; j < PH; ++j) {
        float h = fmaxf(W1[j * 2] * score + W1[j * 2 + 1] * prior + b1[j], 0.0f);
        a2 += W2[j] * h;
      }
      int eid = cand[bs * PC + c];
      link = (eid > 0) ? a2 : -10000.0f;
      out_link[bs * PC + c] = link;
    }
    float mx = wave_max((c < PC) ? link : -1e38f);
    float e  = (c < PC) ? __expf(link - mx) : 0.0f;
    float sm = wave_sum(e);
    if (c < PC) s_norm[c] = e / sm;
  }
  __syncthreads();

  // ---- weighted = sum_c normed[c] * ent_emb[c] ----
  for (int d = tid; d < DENT; d += 256) {
    float a = 0.0f;
    #pragma unroll
    for (int c = 0; c < PC; ++c)
      a += s_norm[c] * s_ent[c * DENT + d];
    out_wt[bs * DENT + d] = a;
  }
}

// =====================================================================
extern "C" void kernel_launch(void* const* d_in, const int* in_sizes, int n_in,
                              void* d_out, int out_size, void* d_ws, size_t ws_size,
                              hipStream_t stream) {
  (void)in_sizes; (void)n_in; (void)out_size; (void)ws_size;

  const float* X      = (const float*)d_in[0];   // (B,T,768)
  // d_in[1] mask: unused by reference
  const int*   spans  = (const int*)d_in[2];     // (B,S,2)
  const int*   cand   = (const int*)d_in[3];     // (B,S,C)
  const float* priors = (const float*)d_in[4];   // (B,S,C)
  // d_in[5] segment ids: unused
  const float* etab   = (const float*)d_in[6];   // (V,300)
  const float* pW     = (const float*)d_in[7];   // (300,768)
  const float* pb     = (const float*)d_in[8];   // (300)
  const float* kg_g   = (const float*)d_in[9];
  const float* kg_b   = (const float*)d_in[10];
  const float* ln_g   = (const float*)d_in[11];
  const float* ln_b   = (const float*)d_in[12];
  const float* attW   = (const float*)d_in[13];  // (1,300)
  const float* attb   = (const float*)d_in[14];  // (1)
  const float* W1     = (const float*)d_in[15];  // (100,2)
  const float* b1     = (const float*)d_in[16];  // (100)
  const float* W2     = (const float*)d_in[17];  // (1,100)
  const float* b2     = (const float*)d_in[18];  // (1)

  const int M = PB * PT;                         // 16384 rows
  float* proj   = (float*)d_ws;                  // M x 304
  float* logits = proj + (size_t)M * DPAD;       // M

  float* out_link = (float*)d_out;                       // (B,S,C)
  float* out_wt   = out_link + (size_t)PB * PS * PC;     // (B,S,300)

  proj_gemm_wmma<<<dim3(M / BM), dim3(256), 0, stream>>>(X, pW, pb, proj);
  attn_logits_kernel<<<dim3(M / 8), dim3(256), 0, stream>>>(proj, attW, attb, logits);
  span_entity_fused<<<dim3(PB * PS), dim3(256), 0, stream>>>(
      proj, logits, spans, cand, priors, etab, kg_g, kg_b, ln_g, ln_b,
      W1, b1, W2, b2, out_link, out_wt);
}
